// SimComE_30932354466027
// MI455X (gfx1250) — matrix operand order repaired and checked by reading, not compiled, not run on hardware
//
#include <hip/hip_runtime.h>
#include <hip/hip_bf16.h>

#define NN 60000
#define EE 240000
#define HH 256
#define MM 64
#define F1C 12
#define F2C 6
#define LL 4
#define NLR 3
#define GG 2000
#define EPSF 1e-5f
#define THREADS 256
#define LDS_PAD 8            // halves of padding per B row -> conflict-free ds_load_b128
#define MAXK 256

typedef __bf16 v16bf __attribute__((ext_vector_type(16)));
typedef __bf16 v8bf  __attribute__((ext_vector_type(8)));
typedef float v8f    __attribute__((ext_vector_type(8)));
typedef int   async_v4i __attribute__((vector_size(16)));   // matches builtin's V4i encoding

union bf16x16 { v16bf v; v8bf h[2]; };

#if defined(__HIP_DEVICE_COMPILE__) && __has_builtin(__builtin_amdgcn_global_load_async_to_lds_b128) && __has_builtin(__builtin_amdgcn_s_wait_asynccnt)
#define HAVE_ASYNC_LDS 1
#else
#define HAVE_ASYNC_LDS 0
#endif

// ---------------- elementwise / gather kernels ----------------

__global__ __launch_bounds__(THREADS) void k_embed(const int* __restrict__ xidx,
                                                   const float* __restrict__ xtab,
                                                   float* __restrict__ x, int n) {
    int id = blockIdx.x * blockDim.x + threadIdx.x;
    if (id >= n * HH) return;
    int node = id / HH, c = id % HH;
    float s = 0.f;
    #pragma unroll
    for (int k = 0; k < 9; ++k) {
        int row = xidx[node * 9 + k];
        s += xtab[(long)row * HH + c];
    }
    x[id] = s;
}

__global__ __launch_bounds__(THREADS) void k_cvt_bf16(const float* __restrict__ in,
                                                      __bf16* __restrict__ out, long n) {
    long id = (long)blockIdx.x * blockDim.x + threadIdx.x;
    if (id >= n) return;
    out[id] = (__bf16)in[id];
}

__global__ __launch_bounds__(THREADS) void k_transpose_cvt(const float* __restrict__ W,
                                                           __bf16* __restrict__ Wt,
                                                           int K, int Ho) {
    int id = blockIdx.x * blockDim.x + threadIdx.x;
    if (id >= K * Ho) return;
    int k = id / Ho, n = id % Ho;
    Wt[(long)n * K + k] = (__bf16)W[(long)k * Ho + n];
}

__global__ __launch_bounds__(THREADS) void k_zero(float* __restrict__ p, long n) {
    long id = (long)blockIdx.x * blockDim.x + threadIdx.x;
    if (id < n) p[id] = 0.f;
}

// ---------------- WMMA GEMM ----------------
// C[rows,Ho] = act(A[rows,K] @ W[K,Ho] + bias + pre) + post
// A: bf16 row-major. Wt: bf16 [Ho,K] (transposed weight).
// Block: 8 waves -> 256 rows x 64 cols. Wave: 32 rows x 64 cols (2x4 16x16 tiles).
// B panel (64 x K) staged once per block into LDS via async-to-LDS.

__global__ __launch_bounds__(THREADS) void k_gemm_bf16(const __bf16* __restrict__ A,
                                                       const __bf16* __restrict__ Wt,
                                                       const float* __restrict__ bias,
                                                       const float* __restrict__ pre,
                                                       const float* __restrict__ post,
                                                       float* __restrict__ C,
                                                       int rows, int K, int Ho, int relu) {
    __shared__ __bf16 ldsB[64 * (MAXK + LDS_PAD)];   // ~33 KB
    const int tid = threadIdx.x;
    const int colBase = blockIdx.x * 64;
    const int kstride = K + LDS_PAD;

    // ---- stage B panel [64 cols x K] into LDS (padded rows) ----
    {
        const __bf16* bsrc = Wt + (size_t)colBase * K;
        int chunksPerCol = K >> 3;                  // 16-byte chunks per column
        int total = chunksPerCol << 6;              // 64 columns
        for (int t = tid; t < total; t += THREADS) {
            int col = t / chunksPerCol;
            int ch  = t - col * chunksPerCol;
            const __bf16* s = bsrc + (size_t)col * K + ch * 8;
            __bf16* d = &ldsB[(size_t)col * kstride + ch * 8];
#if HAVE_ASYNC_LDS
            __builtin_amdgcn_global_load_async_to_lds_b128(
                (__attribute__((address_space(1))) async_v4i*)s,
                (__attribute__((address_space(3))) async_v4i*)d, 0, 0);
#else
            *(v8bf*)d = *(const v8bf*)s;
#endif
        }
#if HAVE_ASYNC_LDS
        __builtin_amdgcn_s_wait_asynccnt(0);
#endif
        __syncthreads();
    }

    const int wave = tid >> 5, lane = tid & 31;
    const int half = lane >> 4, mn = lane & 15;
    long rowBase = ((long)blockIdx.y * 8 + wave) * 32;
    if (rowBase >= rows) return;                    // wave-uniform; after barrier

    // A layout (16-bit 16x32 fragment): lane m=mn; elems 0..7 -> K=half*8+e, 8..15 -> K=16+half*8+(e-8)
    const __bf16* arow0 = A + (rowBase + mn) * (long)K + half * 8;
    const __bf16* arow1 = arow0 + 16 * (long)K;
    // B layout (16-bit 32x16 fragment): lane n=mn; elems 0..7 -> K=half*8+e, 8..15 -> K=16+half*8+(e-8)
    const __bf16* bl = &ldsB[(size_t)mn * kstride + half * 8];

    v8f acc[2][4] = {};
    for (int k0 = 0; k0 < K; k0 += 32) {
        bf16x16 a0, a1;
        a0.h[0] = *(const v8bf*)(arow0 + k0);
        a0.h[1] = *(const v8bf*)(arow0 + k0 + 16);
        a1.h[0] = *(const v8bf*)(arow1 + k0);
        a1.h[1] = *(const v8bf*)(arow1 + k0 + 16);
        #pragma unroll
        for (int ct = 0; ct < 4; ++ct) {
            bf16x16 b;
            const __bf16* bp = bl + ct * 16 * kstride + k0;
            b.h[0] = *(const v8bf*)(bp);
            b.h[1] = *(const v8bf*)(bp + 16);
            acc[0][ct] = __builtin_amdgcn_wmma_f32_16x16x32_bf16(false, a0.v, false, b.v,
                                                                 (short)0, acc[0][ct], false, false);
            acc[1][ct] = __builtin_amdgcn_wmma_f32_16x16x32_bf16(false, a1.v, false, b.v,
                                                                 (short)0, acc[1][ct], false, false);
        }
    }

    #pragma unroll
    for (int rt = 0; rt < 2; ++rt) {
        #pragma unroll
        for (int ct = 0; ct < 4; ++ct) {
            int col = colBase + ct * 16 + mn;
            float bcol = bias ? bias[col] : 0.f;
            #pragma unroll
            for (int r = 0; r < 8; ++r) {
                long row = rowBase + rt * 16 + half * 8 + r;
                long idx = row * Ho + col;
                float v = acc[rt][ct][r] + bcol;
                if (pre) v += pre[idx];
                if (relu) v = fmaxf(v, 0.f);
                if (post) v += post[idx];
                C[idx] = v;
            }
        }
    }
}

// ---------------- small dense GEMM (tiny K: 12 or 6) ----------------

__global__ __launch_bounds__(THREADS) void k_small_gemm(const float* __restrict__ Fin,
                                                        const float* __restrict__ W,
                                                        const float* __restrict__ b,
                                                        float* __restrict__ out,
                                                        int rows, int Kf, int Mo) {
    int id = blockIdx.x * blockDim.x + threadIdx.x;
    if (id >= rows * Mo) return;
    int r = id / Mo, c = id % Mo;
    float s = b[c];
    for (int k = 0; k < Kf; ++k) s += Fin[(long)r * Kf + k] * W[k * Mo + c];
    out[id] = s;
}

// ---------------- BatchNorm (biased var over rows, per column) ----------------

__global__ __launch_bounds__(THREADS) void k_bn_stats(const float* __restrict__ X,
                                                      float* __restrict__ mean,
                                                      float* __restrict__ var,
                                                      int rows, int cols) {
    int c = blockIdx.x;
    float s = 0.f, q = 0.f;
    for (int r = threadIdx.x; r < rows; r += blockDim.x) {
        float v = X[(long)r * cols + c];
        s += v; q += v * v;
    }
    __shared__ float ss[THREADS], sq[THREADS];
    ss[threadIdx.x] = s; sq[threadIdx.x] = q;
    __syncthreads();
    for (int off = THREADS / 2; off > 0; off >>= 1) {
        if (threadIdx.x < off) {
            ss[threadIdx.x] += ss[threadIdx.x + off];
            sq[threadIdx.x] += sq[threadIdx.x + off];
        }
        __syncthreads();
    }
    if (threadIdx.x == 0) {
        float m = ss[0] / rows;
        mean[c] = m;
        var[c] = sq[0] / rows - m * m;
    }
}

__global__ __launch_bounds__(THREADS) void k_bn_apply_relu(const float* __restrict__ X,
                                                           const float* __restrict__ mean,
                                                           const float* __restrict__ var,
                                                           const float* __restrict__ g,
                                                           const float* __restrict__ be,
                                                           float* __restrict__ Y,
                                                           long rows, int cols) {
    long id = (long)blockIdx.x * blockDim.x + threadIdx.x;
    if (id >= rows * cols) return;
    int c = (int)(id % cols);
    float v = (X[id] - mean[c]) * rsqrtf(var[c] + EPSF) * g[c] + be[c];
    Y[id] = fmaxf(v, 0.f);
}

// ---------------- graph conv message + scatter-sum ----------------

__global__ __launch_bounds__(THREADS) void k_message_agg(const float* __restrict__ xl,
                                                         const float* __restrict__ ew,
                                                         const int* __restrict__ src,
                                                         const int* __restrict__ dst,
                                                         float* __restrict__ agg) {
    long id = (long)blockIdx.x * blockDim.x + threadIdx.x;
    if (id >= (long)EE * HH) return;
    int e = (int)(id / HH), c = (int)(id % HH);
    float m = xl[(long)src[e] * HH + c] * ew[id];
    atomicAdd(&agg[(long)dst[e] * HH + c], m);
}

// ---------------- GraphNorm pieces ----------------

__global__ __launch_bounds__(THREADS) void k_count(const int* __restrict__ batch,
                                                   float* __restrict__ cnt, int n) {
    int id = blockIdx.x * blockDim.x + threadIdx.x;
    if (id >= n) return;
    atomicAdd(&cnt[batch[id]], 1.f);
}

__global__ __launch_bounds__(THREADS) void k_seg_accum(const float* __restrict__ h,
                                                       const int* __restrict__ batch,
                                                       float* __restrict__ acc,
                                                       int n, int cols, int square) {
    long id = (long)blockIdx.x * blockDim.x + threadIdx.x;
    if (id >= (long)n * cols) return;
    int node = (int)(id / cols), c = (int)(id % cols);
    float v = h[id];
    if (square) v *= v;
    atomicAdd(&acc[(long)batch[node] * cols + c], v);
}

__global__ __launch_bounds__(THREADS) void k_gn_sub(const float* __restrict__ h,
                                                    const int* __restrict__ batch,
                                                    const float* __restrict__ mean,
                                                    const float* __restrict__ cnt,
                                                    const float* __restrict__ ms,
                                                    float* __restrict__ out,
                                                    int n, int cols) {
    long id = (long)blockIdx.x * blockDim.x + threadIdx.x;
    if (id >= (long)n * cols) return;
    int node = (int)(id / cols), c = (int)(id % cols);
    int b = batch[node];
    float cn = fmaxf(cnt[b], 1.f);
    out[id] = h[id] - ms[c] * (mean[(long)b * cols + c] / cn);
}

__global__ __launch_bounds__(THREADS) void k_gn_apply(const float* __restrict__ out,
                                                      const int* __restrict__ batch,
                                                      const float* __restrict__ var,
                                                      const float* __restrict__ cnt,
                                                      const float* __restrict__ w,
                                                      const float* __restrict__ b,
                                                      float* __restrict__ y,
                                                      int n, int cols) {
    long id = (long)blockIdx.x * blockDim.x + threadIdx.x;
    if (id >= (long)n * cols) return;
    int node = (int)(id / cols), c = (int)(id % cols);
    int g = batch[node];
    float cn = fmaxf(cnt[g], 1.f);
    float v = var[(long)g * cols + c] / cn;
    y[id] = out[id] * rsqrtf(v + EPSF) * w[c] + b[c];
}

// ---------------- host orchestration ----------------

static inline int blks(long n) { return (int)((n + THREADS - 1) / THREADS); }

extern "C" void kernel_launch(void* const* d_in, const int* in_sizes, int n_in,
                              void* d_out, int out_size, void* d_ws, size_t ws_size,
                              hipStream_t stream) {
    const int*   x_idx    = (const int*)d_in[0];
    const int*   eidx     = (const int*)d_in[2];
    const int*   batch    = (const int*)d_in[3];
    const float* feature1 = (const float*)d_in[4];
    const float* feature2 = (const float*)d_in[5];
    const float* x_tab    = (const float*)d_in[6];
    const float* linx_w   = (const float*)d_in[8];
    const float* linx_b   = (const float*)d_in[9];
    const float* linx_g   = (const float*)d_in[10];
    const float* linx_be  = (const float*)d_in[11];
    const float* f1_w1    = (const float*)d_in[12];
    const float* f1_b1    = (const float*)d_in[13];
    const float* f1_g     = (const float*)d_in[14];
    const float* f1_be    = (const float*)d_in[15];
    const float* f1_w2    = (const float*)d_in[16];
    const float* f1_b2    = (const float*)d_in[17];
    const float* f2_w1    = (const float*)d_in[18];
    const float* f2_b1    = (const float*)d_in[19];
    const float* f2_g     = (const float*)d_in[20];
    const float* f2_be    = (const float*)d_in[21];
    const float* f2_w2    = (const float*)d_in[22];
    const float* f2_b2    = (const float*)d_in[23];
    const float* c1_rel_w = (const float*)d_in[24];
    const float* c1_rel_b = (const float*)d_in[25];
    const float* c1_root_w= (const float*)d_in[26];
    const float* c2_rel_w = (const float*)d_in[27];
    const float* c2_rel_b = (const float*)d_in[28];
    const float* c2_root_w= (const float*)d_in[29];
    const float* cat_w    = (const float*)d_in[30];
    const float* cat_b    = (const float*)d_in[31];
    const float* lins_w   = (const float*)d_in[32];
    const float* lins_b   = (const float*)d_in[33];
    const float* gn_w     = (const float*)d_in[34];
    const float* gn_b     = (const float*)d_in[35];
    const float* gn_ms    = (const float*)d_in[36];
    const float* fin_w    = (const float*)d_in[37];
    const float* fin_b    = (const float*)d_in[38];

    const int* src = eidx;
    const int* dst = eidx + EE;

    float* x = (float*)d_out;                     // node features live in d_out

    char* wsp = (char*)d_ws;
    auto alloc = [&](size_t bytes) -> void* {
        void* p = (void*)wsp;
        wsp += (bytes + 255) & ~(size_t)255;
        return p;
    };
    const long NB = (long)NN * HH;
    float*  xl    = (float*)alloc(NB * 4);
    float*  tmp   = (float*)alloc(NB * 4);
    float*  h1    = (float*)alloc(NB * 4);
    float*  h2    = (float*)alloc(NB * 4);
    float*  hA    = (float*)alloc(NB * 4);
    float*  hB    = (float*)alloc(NB * 4);
    float*  agg   = (float*)alloc(NB * 4);
    __bf16* abf   = (__bf16*)alloc(NB * 2);
    __bf16* xlbf  = (__bf16*)alloc(NB * 2);
    float*  ew    = (float*)alloc((long)EE * HH * 4);
    float*  etmp  = (float*)alloc((long)EE * MM * 4);
    __bf16* ebf   = (__bf16*)alloc((long)EE * MM * 2);
    __bf16* wt    = (__bf16*)alloc((size_t)HH * HH * 2);
    float*  bnm   = (float*)alloc(HH * 4);
    float*  bnv   = (float*)alloc(HH * 4);
    float*  gncnt = (float*)alloc(GG * 4);
    float*  gnmean= (float*)alloc((long)GG * HH * 4);
    float*  gnvar = (float*)alloc((long)GG * HH * 4);

    auto gemm = [&](const __bf16* A, const __bf16* Wt, const float* bias,
                    const float* pre, const float* post, float* C,
                    int rows, int K, int Ho, int relu) {
        dim3 grid(Ho / 64, (unsigned)((rows + 255) / 256));
        k_gemm_bf16<<<grid, THREADS, 0, stream>>>(A, Wt, bias, pre, post, C,
                                                  rows, K, Ho, relu);
    };
    auto tcvt = [&](const float* W, int K, int Ho) {
        k_transpose_cvt<<<blks((long)K * Ho), THREADS, 0, stream>>>(W, wt, K, Ho);
    };
    auto cvt = [&](const float* in, __bf16* out, long n) {
        k_cvt_bf16<<<blks(n), THREADS, 0, stream>>>(in, out, n);
    };

    // x = x_tab[x_idx].sum(1)
    k_embed<<<blks(NB), THREADS, 0, stream>>>(x_idx, x_tab, x, NN);

    for (int l = 0; l < LL; ++l) {
        // ---- lin_x: relu(BN(x @ W + b)) ----
        cvt(x, abf, NB);
        tcvt(linx_w + (long)l * HH * HH, HH, HH);
        gemm(abf, wt, linx_b + l * HH, nullptr, nullptr, tmp, NN, HH, HH, 0);
        k_bn_stats<<<HH, THREADS, 0, stream>>>(tmp, bnm, bnv, NN, HH);
        k_bn_apply_relu<<<blks(NB), THREADS, 0, stream>>>(tmp, bnm, bnv,
            linx_g + l * HH, linx_be + l * HH, xl, NN, HH);
        cvt(xl, xlbf, NB);

        // ---- two edge-MLP + conv branches ----
        for (int br = 0; br < 2; ++br) {
            const float* fw1 = br ? f2_w1 : f1_w1;  const float* fb1 = br ? f2_b1 : f1_b1;
            const float* fg  = br ? f2_g  : f1_g;   const float* fbe = br ? f2_be : f1_be;
            const float* fw2 = br ? f2_w2 : f1_w2;  const float* fb2 = br ? f2_b2 : f1_b2;
            const float* ft  = br ? feature2 : feature1;
            int Fc = br ? F2C : F1C;
            const float* relw  = br ? c2_rel_w  : c1_rel_w;
            const float* relb  = br ? c2_rel_b  : c1_rel_b;
            const float* rootw = br ? c2_root_w : c1_root_w;
            float* hout = br ? h2 : h1;

            // edge weight MLP: relu(relu(BN(f@W1+b1)) @ W2 + b2)
            k_small_gemm<<<blks((long)EE * MM), THREADS, 0, stream>>>(
                ft, fw1 + (long)l * Fc * MM, fb1 + l * MM, etmp, EE, Fc, MM);
            k_bn_stats<<<MM, THREADS, 0, stream>>>(etmp, bnm, bnv, EE, MM);
            k_bn_apply_relu<<<blks((long)EE * MM), THREADS, 0, stream>>>(
                etmp, bnm, bnv, fg + l * MM, fbe + l * MM, etmp, EE, MM);
            cvt(etmp, ebf, (long)EE * MM);
            tcvt(fw2 + (long)l * MM * HH, MM, HH);
            gemm(ebf, wt, fb2 + l * HH, nullptr, nullptr, ew, EE, MM, HH, 1);

            // aggregate messages: agg[dst] += xl[src] * ew
            k_zero<<<blks(NB), THREADS, 0, stream>>>(agg, NB);
            k_message_agg<<<blks((long)EE * HH), THREADS, 0, stream>>>(xl, ew, src, dst, agg);

            // h = relu(agg @ rel_w + rel_b + xl @ root_w)
            cvt(agg, abf, NB);
            tcvt(relw + (long)l * HH * HH, HH, HH);
            gemm(abf, wt, relb + l * HH, nullptr, nullptr, tmp, NN, HH, HH, 0);
            tcvt(rootw + (long)l * HH * HH, HH, HH);
            gemm(xlbf, wt, nullptr, tmp, nullptr, hout, NN, HH, HH, 1);
        }

        // ---- h = relu([h1|h2] @ cat_w + cat_b) + xl ----
        cvt(h1, abf, NB);
        tcvt(cat_w + (long)l * 2 * HH * HH, HH, HH);
        gemm(abf, wt, cat_b + l * HH, nullptr, nullptr, tmp, NN, HH, HH, 0);
        cvt(h2, abf, NB);
        tcvt(cat_w + (long)l * 2 * HH * HH + (long)HH * HH, HH, HH);
        gemm(abf, wt, nullptr, tmp, xl, hA, NN, HH, HH, 1);

        // ---- residual linears: h = relu(h @ W + b) + h ----
        float* cur = hA; float* alt = hB;
        for (int k = 0; k < NLR; ++k) {
            cvt(cur, abf, NB);
            tcvt(lins_w + ((long)l * NLR + k) * HH * HH, HH, HH);
            gemm(abf, wt, lins_b + ((long)l * NLR + k) * HH, nullptr, cur, alt, NN, HH, HH, 1);
            float* t = cur; cur = alt; alt = t;
        }

        // ---- GraphNorm ----
        k_zero<<<blks(GG), THREADS, 0, stream>>>(gncnt, GG);
        k_zero<<<blks((long)GG * HH), THREADS, 0, stream>>>(gnmean, (long)GG * HH);
        k_count<<<blks(NN), THREADS, 0, stream>>>(batch, gncnt, NN);
        k_seg_accum<<<blks(NB), THREADS, 0, stream>>>(cur, batch, gnmean, NN, HH, 0);
        k_gn_sub<<<blks(NB), THREADS, 0, stream>>>(cur, batch, gnmean, gncnt,
                                                   gn_ms + l * HH, alt, NN, HH);
        k_zero<<<blks((long)GG * HH), THREADS, 0, stream>>>(gnvar, (long)GG * HH);
        k_seg_accum<<<blks(NB), THREADS, 0, stream>>>(alt, batch, gnvar, NN, HH, 1);
        k_gn_apply<<<blks(NB), THREADS, 0, stream>>>(alt, batch, gnvar, gncnt,
                                                     gn_w + l * HH, gn_b + l * HH, cur, NN, HH);

        // ---- final linear of block: x = h @ fin_w + fin_b ----
        cvt(cur, abf, NB);
        tcvt(fin_w + (long)l * HH * HH, HH, HH);
        gemm(abf, wt, fin_b + l * HH, nullptr, nullptr, x, NN, HH, HH, 0);
    }
}
